// CausalSelfAttention_70342974374189
// MI455X (gfx1250) — compile-verified, matrix operand order checked by reading, not run on hardware
//
#include <hip/hip_runtime.h>

typedef __attribute__((ext_vector_type(16))) __bf16 v16bf;
typedef __attribute__((ext_vector_type(8)))  __bf16 v8bf;
typedef __attribute__((ext_vector_type(4)))  __bf16 v4bf;
typedef __attribute__((ext_vector_type(8)))  float  v8f;
typedef __attribute__((ext_vector_type(4)))  float  v4f;

#define B_   2
#define T_   2048
#define C_   2048
#define NH   16
#define NKV  4
#define HD   128
#define KVC  512   /* NKV*HD */

// Async global->LDS copy via inline asm (bypasses builtin signature issues;
// CDNA5 ISA: VDST = LDS byte address VGPR, VADDR =
// 64-bit global address, SADDR = off, tracked by ASYNCcnt).
#define ASYNC_COPY_B128(ldsptr, gptr)                                          \
  asm volatile("global_load_async_to_lds_b128 %0, %1, off"                     \
               :                                                               \
               : "v"((unsigned)(size_t)(ldsptr)), "v"((gptr))                  \
               : "memory")
#define WAIT_ASYNC() asm volatile("s_wait_asynccnt 0" ::: "memory")

// ---------------------------------------------------------------------------
// Fragment loader: A/B operand for v_wmma_f32_16x16x32_bf16.
// 16-bit A-matrix 16x32 layout (ISA 7.12.2): lane = half*16 + m,
//   VGPRs 0-3 hold K = half*8 .. half*8+7, VGPRs 4-7 hold K = 16+half*8 ..
// B fragments use the identical pattern when B^T (N x K) is stored row-major.
// ---------------------------------------------------------------------------
__device__ __forceinline__ v16bf load_frag(const __bf16* base, int row, int kbase,
                                           int ld, int lane) {
  const int m    = lane & 15;
  const int half = lane >> 4;
  const __bf16* p = base + (size_t)(row + m) * ld + kbase + half * 8;
  v8bf lo = *(const v8bf*)(p);
  v8bf hi = *(const v8bf*)(p + 16);
  return __builtin_shufflevector(lo, hi, 0,1,2,3,4,5,6,7,8,9,10,11,12,13,14,15);
}

__device__ __forceinline__ v8f wmma_bf16(v16bf a, v16bf b, v8f c) {
  return __builtin_amdgcn_wmma_f32_16x16x32_bf16(false, a, false, b,
                                                 (short)0, c, false, false);
}

// ---------------------------------------------------------------------------
// Generic GEMM: C[M,N] (f32) = A[M,K] (TA) * W[N,K]^T (f32 weights, row major)
// Block tile 128x128, 8 waves, each wave computes 64x32 as 4x2 WMMA frags.
// ---------------------------------------------------------------------------
template <typename TA>
__global__ __launch_bounds__(256)
void gemm_wmma_bf16(const TA* __restrict__ A, const float* __restrict__ W,
                    float* __restrict__ C, int M, int N, int K) {
  __shared__ __bf16 sA[128][40];   // 32 + 8 pad, rows 16B aligned
  __shared__ __bf16 sB[128][40];

  const int tid  = threadIdx.x;
  const int lane = tid & 31;
  const int w    = tid >> 5;
  const int wm   = w & 1;          // 2 wave rows   (64 M each)
  const int wn   = w >> 1;         // 4 wave cols   (32 N each)
  const int mb   = blockIdx.y * 128;
  const int nb   = blockIdx.x * 128;

  v8f acc[4][2];
#pragma unroll
  for (int i = 0; i < 4; ++i)
#pragma unroll
    for (int j = 0; j < 2; ++j) acc[i][j] = (v8f){0,0,0,0,0,0,0,0};

  for (int kb = 0; kb < K; kb += 32) {
    // cooperative stage: 128x32 of A and of W -> bf16 (packed 8B DS stores)
#pragma unroll
    for (int p = 0; p < 4; ++p) {
      int e   = p * 1024 + tid * 4;
      int row = e >> 5;
      int col = e & 31;
      const TA*    ap = A + (size_t)(mb + row) * K + kb + col;
      const float* bp = W + (size_t)(nb + row) * K + kb + col;
      v4bf a4;
      if constexpr (sizeof(TA) == 2) {
        a4 = *(const v4bf*)ap;                 // already bf16: straight copy
      } else {
        v4f av = *(const v4f*)ap;
#pragma unroll
        for (int j = 0; j < 4; ++j) a4[j] = (__bf16)av[j];
      }
      v4f bv = *(const v4f*)bp;
      v4bf b4;
#pragma unroll
      for (int j = 0; j < 4; ++j) b4[j] = (__bf16)bv[j];
      *(v4bf*)&sA[row][col] = a4;
      *(v4bf*)&sB[row][col] = b4;
    }
    __syncthreads();

    v16bf af[4], bfr[2];
#pragma unroll
    for (int i = 0; i < 4; ++i) af[i]  = load_frag(&sA[0][0], wm * 64 + i * 16, 0, 40, lane);
#pragma unroll
    for (int j = 0; j < 2; ++j) bfr[j] = load_frag(&sB[0][0], wn * 32 + j * 16, 0, 40, lane);
#pragma unroll
    for (int i = 0; i < 4; ++i)
#pragma unroll
      for (int j = 0; j < 2; ++j)
        acc[i][j] = wmma_bf16(af[i], bfr[j], acc[i][j]);
    __syncthreads();
  }

  // D layout (ISA 7.12.2): vgpr r, lanes 0-15 -> M=r, lanes 16-31 -> M=8+r; N=lane&15
  const int nn = lane & 15, half = lane >> 4;
#pragma unroll
  for (int i = 0; i < 4; ++i)
#pragma unroll
    for (int j = 0; j < 2; ++j) {
      int n  = nb + wn * 32 + j * 16 + nn;
      int m0 = mb + wm * 64 + i * 16 + half * 8;
#pragma unroll
      for (int r = 0; r < 8; ++r) C[(size_t)(m0 + r) * N + n] = acc[i][j][r];
    }
}

// ---------------------------------------------------------------------------
// RoPE (adjacent-pair rotation) fp32 -> bf16.  One thread per (row, head, i).
// ---------------------------------------------------------------------------
__global__ void rope_to_bf16(const float* __restrict__ in, __bf16* __restrict__ out,
                             const float* __restrict__ cs, const float* __restrict__ sn,
                             int nheads, long total) {
  long idx = (long)blockIdx.x * blockDim.x + threadIdx.x;
  if (idx >= total) return;
  int  i    = (int)(idx & 63);
  long rest = idx >> 6;
  int  hh   = (int)(rest % nheads);
  long row  = rest / nheads;
  int  t    = (int)(row % T_);
  float c = cs[t * 64 + i];
  float s = sn[t * 64 + i];
  long base = row * (long)(nheads * HD) + hh * HD + 2 * i;
  float t0 = in[base], t1 = in[base + 1];
  out[base]     = (__bf16)(t0 * c - t1 * s);
  out[base + 1] = (__bf16)(t0 * s + t1 * c);
}

__global__ void f32_to_bf16(const float* __restrict__ in, __bf16* __restrict__ out, long n) {
  long i = (long)blockIdx.x * blockDim.x + threadIdx.x;
  if (i < n) out[i] = (__bf16)in[i];
}

// ---------------------------------------------------------------------------
// Flash attention: block = (b, h, 128-row q tile). 8 waves x 16 q rows.
// Key tile = 64. K staged row-major in LDS via async-to-LDS copies, V staged
// transposed so both QK^T and PV use the same B-fragment load pattern.
// Online softmax in fp32.
// ---------------------------------------------------------------------------
#define KT 64

__global__ __launch_bounds__(256)
void attn_kernel(const __bf16* __restrict__ Q, const __bf16* __restrict__ Kb,
                 const __bf16* __restrict__ Vb, __bf16* __restrict__ O) {
  __shared__ __bf16 sK [KT][136];     // [key][d],  128 + 8 pad   (17408 B)
  __shared__ __bf16 sVt[128][72];     // [d][key],  KT + 8 pad    (18432 B)
  __shared__ __bf16 sP [8][16][72];   // per-wave P staging       (18432 B)

  const int tid  = threadIdx.x;
  const int lane = tid & 31;
  const int w    = tid >> 5;
  const int qt   = blockIdx.x & 15;
  const int h    = (blockIdx.x >> 4) & 15;
  const int b    = blockIdx.x >> 8;
  const int kvh  = h >> 2;                 // GQA: 4 Q heads share one KV head
  const long rowQ  = (long)b * T_ + qt * 128;
  const long rowKV = (long)b * T_;
  const float scale = 0.08838834764831845f;  // 1/sqrt(128)

  // Load this wave's Q (16 rows x 128) as 4 A-fragments, pre-scaled.
  v16bf qf[4];
  {
    const int m = lane & 15, half = lane >> 4;
    const __bf16* qp = Q + (rowQ + w * 16 + m) * C_ + h * HD;
#pragma unroll
    for (int c = 0; c < 4; ++c) {
      v8bf lo = *(const v8bf*)(qp + c * 32 + half * 8);
      v8bf hi = *(const v8bf*)(qp + c * 32 + half * 8 + 16);
      v16bf f = __builtin_shufflevector(lo, hi, 0,1,2,3,4,5,6,7,8,9,10,11,12,13,14,15);
#pragma unroll
      for (int e = 0; e < 16; ++e) f[e] = (__bf16)((float)f[e] * scale);
      qf[c] = f;
    }
  }

  v8f o[8];
#pragma unroll
  for (int d = 0; d < 8; ++d) o[d] = (v8f){0,0,0,0,0,0,0,0};
  float rmax[8], rsum[8];
#pragma unroll
  for (int r = 0; r < 8; ++r) { rmax[r] = -1e30f; rsum[r] = 0.f; }

  const int rr = tid >> 4;             // 0..15
  const int c8 = (tid & 15) * 8;       // 0..120

  const int ntiles = qt * 2 + 2;       // causal: keys <= (qt+1)*128
  for (int kt = 0; kt < ntiles; ++kt) {
    // ---- stage K tile (async copy) and V tile (transposed, VALU scatter)
#pragma unroll
    for (int p = 0; p < 4; ++p) {
      int key = p * 16 + rr;
      const __bf16* kp = Kb + (rowKV + kt * KT + key) * KVC + kvh * HD + c8;
      ASYNC_COPY_B128(&sK[key][c8], kp);
      const __bf16* vp = Vb + (rowKV + kt * KT + key) * KVC + kvh * HD + c8;
      v8bf vv = *(const v8bf*)vp;
#pragma unroll
      for (int j = 0; j < 8; ++j) sVt[c8 + j][key] = vv[j];
    }
    // prefetch next K tile into L2 while we compute on this one
    if (kt + 1 < ntiles) {
      const __bf16* nk = Kb + (rowKV + (kt + 1) * KT + rr) * KVC + kvh * HD + c8;
      __builtin_prefetch(nk, 0, 1);
    }
    WAIT_ASYNC();
    __syncthreads();

    // ---- scores S = Q K^T  (16 x 64 per wave, four 16-col chunks)
    v8f sfr[4];
#pragma unroll
    for (int nc = 0; nc < 4; ++nc) {
      v8f s = (v8f){0,0,0,0,0,0,0,0};
#pragma unroll
      for (int c = 0; c < 4; ++c) {
        v16bf kf = load_frag(&sK[0][0], nc * 16, c * 32, 136, lane);
        s = wmma_bf16(qf[c], kf, s);
      }
      // causal mask in D-layout: n = lane&15, m = r + 8*half
      int kglob = kt * KT + nc * 16 + (lane & 15);
#pragma unroll
      for (int r = 0; r < 8; ++r) {
        int qrow = qt * 128 + w * 16 + r + 8 * (lane >> 4);
        if (kglob > qrow) s[r] = -1e30f;
      }
      sfr[nc] = s;
    }

    // ---- online softmax (row = r + 8*half lives on 16 lanes of a half-wave)
#pragma unroll
    for (int r = 0; r < 8; ++r) {
      float v = fmaxf(fmaxf(sfr[0][r], sfr[1][r]), fmaxf(sfr[2][r], sfr[3][r]));
#pragma unroll
      for (int mk = 1; mk < 16; mk <<= 1) v = fmaxf(v, __shfl_xor(v, mk, 16));
      float nm    = fmaxf(rmax[r], v);
      float alpha = __expf(rmax[r] - nm);
      rmax[r] = nm;
      float pe[4], ps = 0.f;
#pragma unroll
      for (int nc = 0; nc < 4; ++nc) { pe[nc] = __expf(sfr[nc][r] - nm); ps += pe[nc]; }
#pragma unroll
      for (int mk = 1; mk < 16; mk <<= 1) ps += __shfl_xor(ps, mk, 16);
      rsum[r] = rsum[r] * alpha + ps;
#pragma unroll
      for (int d = 0; d < 8; ++d) o[d][r] = o[d][r] * alpha;
      int m  = r + 8 * (lane >> 4);
      int kc = lane & 15;
#pragma unroll
      for (int nc = 0; nc < 4; ++nc) sP[w][m][nc * 16 + kc] = (__bf16)pe[nc];
    }
    __syncthreads();

    // ---- O += P V   (P: 16x64 -> two A-fragments; V^T rows give B-fragments)
    v16bf pa0 = load_frag(&sP[w][0][0], 0,  0, 72, lane);
    v16bf pa1 = load_frag(&sP[w][0][0], 0, 32, 72, lane);
#pragma unroll
    for (int d = 0; d < 8; ++d) {
      v16bf vf0 = load_frag(&sVt[0][0], d * 16,  0, 72, lane);
      v16bf vf1 = load_frag(&sVt[0][0], d * 16, 32, 72, lane);
      o[d] = wmma_bf16(pa0, vf0, o[d]);
      o[d] = wmma_bf16(pa1, vf1, o[d]);
    }
    __syncthreads();
  }

  // ---- normalize and write attention output (bf16, [B*T, C], col = h*128+d)
  const int nn = lane & 15, half = lane >> 4;
#pragma unroll
  for (int d = 0; d < 8; ++d)
#pragma unroll
    for (int r = 0; r < 8; ++r) {
      float val = o[d][r] / rsum[r];
      long  row = rowQ + w * 16 + r + 8 * half;
      O[row * C_ + h * HD + d * 16 + nn] = (__bf16)val;
    }
}

// ---------------------------------------------------------------------------
extern "C" void kernel_launch(void* const* d_in, const int* in_sizes, int n_in,
                              void* d_out, int out_size, void* d_ws, size_t ws_size,
                              hipStream_t stream) {
  const float* x  = (const float*)d_in[0];
  const float* fc = (const float*)d_in[1];
  const float* fs = (const float*)d_in[2];
  const float* wq = (const float*)d_in[3];
  const float* wk = (const float*)d_in[4];
  const float* wv = (const float*)d_in[5];
  const float* wo = (const float*)d_in[6];
  float* out = (float*)d_out;

  const long Mrows = (long)B_ * T_;   // 4096
  char* p = (char*)d_ws;
  float*  Qf = (float*)p;  p += Mrows * C_  * sizeof(float);
  float*  Kf = (float*)p;  p += Mrows * KVC * sizeof(float);
  float*  Vf = (float*)p;  p += Mrows * KVC * sizeof(float);
  __bf16* Qb = (__bf16*)p; p += Mrows * C_  * sizeof(__bf16);
  __bf16* Kb = (__bf16*)p; p += Mrows * KVC * sizeof(__bf16);
  __bf16* Vb = (__bf16*)p; p += Mrows * KVC * sizeof(__bf16);
  __bf16* Sb = (__bf16*)p; p += Mrows * C_  * sizeof(__bf16);

  dim3 blk(256);

  // QKV projections (WMMA bf16, f32 accumulate)
  gemm_wmma_bf16<float><<<dim3(C_ / 128, Mrows / 128), blk, 0, stream>>>(x, wq, Qf, (int)Mrows, C_, C_);
  gemm_wmma_bf16<float><<<dim3(KVC / 128, Mrows / 128), blk, 0, stream>>>(x, wk, Kf, (int)Mrows, KVC, C_);
  gemm_wmma_bf16<float><<<dim3(KVC / 128, Mrows / 128), blk, 0, stream>>>(x, wv, Vf, (int)Mrows, KVC, C_);

  // RoPE + downcast to bf16
  long tq = Mrows * NH  * 64;
  long tk = Mrows * NKV * 64;
  long tv = Mrows * KVC;
  rope_to_bf16<<<(unsigned)((tq + 255) / 256), blk, 0, stream>>>(Qf, Qb, fc, fs, NH,  tq);
  rope_to_bf16<<<(unsigned)((tk + 255) / 256), blk, 0, stream>>>(Kf, Kb, fc, fs, NKV, tk);
  f32_to_bf16 <<<(unsigned)((tv + 255) / 256), blk, 0, stream>>>(Vf, Vb, tv);

  // Fused causal flash attention (WMMA for QK^T and PV)
  attn_kernel<<<B_ * NH * (T_ / 128), blk, 0, stream>>>(Qb, Kb, Vb, Sb);

  // Output projection
  gemm_wmma_bf16<__bf16><<<dim3(C_ / 128, Mrows / 128), blk, 0, stream>>>(Sb, wo, out, (int)Mrows, C_, C_);
}
